// Block_1726576855578
// MI455X (gfx1250) — matrix operand order repaired and checked by reading, not compiled, not run on hardware
//
#include <hip/hip_runtime.h>
#include <stdint.h>
#include <stddef.h>

// Problem constants (match reference)
#define B_SZ    4096
#define D_SZ    2048
#define MM_SZ   1600
#define NCHUNK  20
#define RANK_N  15
#define OUT_N   3000
#define CSIZE   80     // SIZE
#define TDIM    1200   // SIZE*RANK

typedef __attribute__((ext_vector_type(16))) __bf16        v16bf;
typedef __attribute__((ext_vector_type(8)))  float         v8f;
typedef __attribute__((ext_vector_type(4)))  unsigned int  v4u;

union OpPack { v4u q[2]; v16bf v; };

__device__ __forceinline__ unsigned short f2bf(float f) {
  union { float f; unsigned u; } c; c.f = f;
  unsigned u = c.u + 0x7FFFu + ((c.u >> 16) & 1u);   // round-to-nearest-even
  return (unsigned short)(u >> 16);
}
__device__ __forceinline__ float bf2f(unsigned short h) {
  union { unsigned u; float f; } c; c.u = ((unsigned)h) << 16;
  return c.f;
}

// ---------------------------------------------------------------------------
// fp32 -> (bf16 hi, bf16 lo) split, 4 elements/thread, vectorized
// ---------------------------------------------------------------------------
__global__ void k_split(const float* __restrict__ in,
                        unsigned short* __restrict__ hi,
                        unsigned short* __restrict__ lo, int n4) {
  int i = blockIdx.x * blockDim.x + threadIdx.x;
  if (i >= n4) return;
  float4 f = ((const float4*)in)[i];
  float fv[4] = {f.x, f.y, f.z, f.w};
  unsigned short hs[4], ls[4];
#pragma unroll
  for (int j = 0; j < 4; j++) {
    hs[j] = f2bf(fv[j]);
    ls[j] = f2bf(fv[j] - bf2f(hs[j]));
  }
  ((uint2*)hi)[i] = make_uint2(hs[0] | ((unsigned)hs[1] << 16),
                               hs[2] | ((unsigned)hs[3] << 16));
  ((uint2*)lo)[i] = make_uint2(ls[0] | ((unsigned)ls[1] << 16),
                               ls[2] | ((unsigned)ls[3] << 16));
}

// ---------------------------------------------------------------------------
// Split-bf16 GEMM: C[M,N] = A[M,K] @ W[N,K]^T + bias
// A,W given as bf16 hi/lo pairs. OUTMODE 0: fp32 C. OUTMODE 1: hi/lo bf16 C.
// WG tile 128x128, BK=32, 8 waves (wave tile 32Mx64N -> 2x4 accum tiles).
//
// Operand layouts (CDNA5 ISA 7.12.2/7.12.4, wave32, 16-bit, 16x16x32):
//   A (16x32): lane<16 -> M=lane, K in {0..7} U {16..23};
//              lane>=16 -> M=lane-16, K in {8..15} U {24..31}  (paired runs)
//   B (32x16): lane&15 = N (a row of W); lane<16 holds K {0..7} U {16..23},
//              lane>=16 holds K {8..15} U {24..31} (same as A).
//   C/D:       VGPR i -> row (i + 8*(lane>=16)), col = lane&15.
// ---------------------------------------------------------------------------
template <int OUTMODE>
__global__ __launch_bounds__(256)
void k_gemm(const unsigned short* __restrict__ Ah, const unsigned short* __restrict__ Al,
            const unsigned short* __restrict__ Bh, const unsigned short* __restrict__ Bl,
            const float* __restrict__ bias,
            float* __restrict__ Cf,
            unsigned short* __restrict__ Ch, unsigned short* __restrict__ Cl,
            int N, int K) {
  const int LDK = 40;  // 32 + 8 halves pad: conflict-free ds_load_b128
  __shared__ __align__(16) unsigned short sAh[128 * LDK], sAl[128 * LDK];
  __shared__ __align__(16) unsigned short sBh[128 * LDK], sBl[128 * LDK];

  const int tid  = threadIdx.x;
  const int lane = tid & 31;
  const int wave = tid >> 5;
  const int wm   = wave & 3;       // 4 waves along M
  const int wn   = wave >> 2;      // 2 waves along N
  const int m0   = blockIdx.y * 128;
  const int n0   = blockIdx.x * 128;
  const int lrow = lane & 15;
  const int hsel = lane >> 4;
  const int kbA  = hsel * 8;       // A-operand K base within paired runs
  const int kbB  = hsel * 8;       // B-operand K base within paired runs

  const v8f VZ = {0.f, 0.f, 0.f, 0.f, 0.f, 0.f, 0.f, 0.f};
  const v4u QZ = {0u, 0u, 0u, 0u};
  v8f acc[2][4];
#pragma unroll
  for (int x = 0; x < 2; x++)
#pragma unroll
    for (int y = 0; y < 4; y++) acc[x][y] = VZ;

  for (int k0 = 0; k0 < K; k0 += 32) {
    __syncthreads();
#pragma unroll
    for (int j = 0; j < 2; j++) {
      int slot = tid + j * 256;
      int row  = slot >> 2;
      int kh   = (slot & 3) * 8;
      *(v4u*)(sAh + row * LDK + kh) = *(const v4u*)(Ah + (size_t)(m0 + row) * K + k0 + kh);
      *(v4u*)(sAl + row * LDK + kh) = *(const v4u*)(Al + (size_t)(m0 + row) * K + k0 + kh);
      int nr = n0 + row;
      if (nr < N) {
        *(v4u*)(sBh + row * LDK + kh) = *(const v4u*)(Bh + (size_t)nr * K + k0 + kh);
        *(v4u*)(sBl + row * LDK + kh) = *(const v4u*)(Bl + (size_t)nr * K + k0 + kh);
      } else {
        *(v4u*)(sBh + row * LDK + kh) = QZ;
        *(v4u*)(sBl + row * LDK + kh) = QZ;
      }
    }
    __syncthreads();

    OpPack pa[2][2], pb[4][2];
#pragma unroll
    for (int im = 0; im < 2; im++) {
      int r = wm * 32 + im * 16 + lrow;
      pa[im][0].q[0] = *(const v4u*)(sAh + r * LDK + kbA);
      pa[im][0].q[1] = *(const v4u*)(sAh + r * LDK + kbA + 16);
      pa[im][1].q[0] = *(const v4u*)(sAl + r * LDK + kbA);
      pa[im][1].q[1] = *(const v4u*)(sAl + r * LDK + kbA + 16);
    }
#pragma unroll
    for (int in = 0; in < 4; in++) {
      int r = wn * 64 + in * 16 + lrow;
      pb[in][0].q[0] = *(const v4u*)(sBh + r * LDK + kbB);
      pb[in][0].q[1] = *(const v4u*)(sBh + r * LDK + kbB + 16);
      pb[in][1].q[0] = *(const v4u*)(sBl + r * LDK + kbB);
      pb[in][1].q[1] = *(const v4u*)(sBl + r * LDK + kbB + 16);
    }
#pragma unroll
    for (int im = 0; im < 2; im++) {
#pragma unroll
      for (int in = 0; in < 4; in++) {
        acc[im][in] = __builtin_amdgcn_wmma_f32_16x16x32_bf16(
            false, pa[im][0].v, false, pb[in][0].v, (short)0, acc[im][in], false, false);
        acc[im][in] = __builtin_amdgcn_wmma_f32_16x16x32_bf16(
            false, pa[im][0].v, false, pb[in][1].v, (short)0, acc[im][in], false, false);
        acc[im][in] = __builtin_amdgcn_wmma_f32_16x16x32_bf16(
            false, pa[im][1].v, false, pb[in][0].v, (short)0, acc[im][in], false, false);
      }
    }
  }

  // Epilogue. C layout: VGPR i -> row (i + hsel*8), col = lane&15 within tile.
#pragma unroll
  for (int im = 0; im < 2; im++) {
    int rb = m0 + wm * 32 + im * 16 + hsel * 8;
#pragma unroll
    for (int in = 0; in < 4; in++) {
      int col = n0 + wn * 64 + in * 16 + lrow;
      if (col < N) {
        float bv = bias ? bias[col] : 0.f;
#pragma unroll
        for (int i = 0; i < 8; i++) {
          float v = acc[im][in][i] + bv;
          size_t off = (size_t)(rb + i) * N + col;
          if (OUTMODE == 0) {
            Cf[off] = v;
          } else {
            unsigned short h = f2bf(v);
            Ch[off] = h;
            Cl[off] = f2bf(v - bf2f(h));
          }
        }
      }
    }
  }
}

// ---------------------------------------------------------------------------
// Fused bilinear stage: per wave = 16 batch rows x one chunk.
// y0/y1 tiles (K=80, padded to 96) built with WMMA; z = sum_r y0*y1;
// signed-sqrt; per-(row,chunk) L2 normalize over 80 cols; store z hi/lo bf16.
// ---------------------------------------------------------------------------
__global__ __launch_bounds__(256)
void k_bilinear(const unsigned short* __restrict__ h0h, const unsigned short* __restrict__ h0l,
                const unsigned short* __restrict__ h1h, const unsigned short* __restrict__ h1l,
                const unsigned short* __restrict__ mW0h, const unsigned short* __restrict__ mW0l,
                const float* __restrict__ mb0,
                const unsigned short* __restrict__ mW1h, const unsigned short* __restrict__ mW1l,
                const float* __restrict__ mb1,
                unsigned short* __restrict__ zh, unsigned short* __restrict__ zl) {
  const int tid     = threadIdx.x;
  const int lane    = tid & 31;
  const int wave    = tid >> 5;
  const int c       = blockIdx.y;
  const int rowbase = blockIdx.x * 128 + wave * 16;
  const int lrow    = lane & 15;
  const int hsel    = lane >> 4;
  const int kbA     = hsel * 8;
  const int kbB     = hsel * 8;
  const v4u QZ = {0u, 0u, 0u, 0u};
  const v8f VZ = {0.f, 0.f, 0.f, 0.f, 0.f, 0.f, 0.f, 0.f};

  // A operands (h tiles), reused across all RANK x 5 y-tiles.
  // K padded 80 -> 96: for kc==2 the second paired run (K 80..95) is zero.
  OpPack A0[3][2], A1[3][2];
  const size_t arow = (size_t)(rowbase + lrow) * MM_SZ + (size_t)c * CSIZE;
#pragma unroll
  for (int kc = 0; kc < 3; kc++) {
    int ko = kc * 32 + kbA;
    A0[kc][0].q[0] = *(const v4u*)(h0h + arow + ko);
    A0[kc][1].q[0] = *(const v4u*)(h0l + arow + ko);
    A1[kc][0].q[0] = *(const v4u*)(h1h + arow + ko);
    A1[kc][1].q[0] = *(const v4u*)(h1l + arow + ko);
    if (kc < 2) {
      A0[kc][0].q[1] = *(const v4u*)(h0h + arow + ko + 16);
      A0[kc][1].q[1] = *(const v4u*)(h0l + arow + ko + 16);
      A1[kc][0].q[1] = *(const v4u*)(h1h + arow + ko + 16);
      A1[kc][1].q[1] = *(const v4u*)(h1l + arow + ko + 16);
    } else {
      A0[kc][0].q[1] = QZ; A0[kc][1].q[1] = QZ;
      A1[kc][0].q[1] = QZ; A1[kc][1].q[1] = QZ;
    }
  }

  v8f zacc[5];
#pragma unroll
  for (int n = 0; n < 5; n++) zacc[n] = VZ;

  for (int r = 0; r < RANK_N; r++) {
#pragma unroll
    for (int n = 0; n < 5; n++) {
      const int t = r * CSIZE + n * 16 + lrow;   // mW row (= B column) for this lane
      const size_t brow = ((size_t)c * TDIM + t) * CSIZE;
      // B operand K: paired runs [kc*32 + 8*hsel, +8) U [kc*32 + 16 + 8*hsel, +8).
      // For kc==2 the second run is K 80..95 -> padding -> zero.
      v8f y0, y1;
      {
        float bv = mb0[c * TDIM + t];
#pragma unroll
        for (int i = 0; i < 8; i++) y0[i] = bv;
      }
#pragma unroll
      for (int kc = 0; kc < 3; kc++) {
        OpPack bh, bl;
        int ko = kc * 32 + kbB;
        bh.q[0] = *(const v4u*)(mW0h + brow + ko);
        bl.q[0] = *(const v4u*)(mW0l + brow + ko);
        if (kc < 2) {
          bh.q[1] = *(const v4u*)(mW0h + brow + ko + 16);
          bl.q[1] = *(const v4u*)(mW0l + brow + ko + 16);
        } else {
          bh.q[1] = QZ; bl.q[1] = QZ;
        }
        y0 = __builtin_amdgcn_wmma_f32_16x16x32_bf16(false, A0[kc][0].v, false, bh.v, (short)0, y0, false, false);
        y0 = __builtin_amdgcn_wmma_f32_16x16x32_bf16(false, A0[kc][0].v, false, bl.v, (short)0, y0, false, false);
        y0 = __builtin_amdgcn_wmma_f32_16x16x32_bf16(false, A0[kc][1].v, false, bh.v, (short)0, y0, false, false);
      }
      {
        float bv = mb1[c * TDIM + t];
#pragma unroll
        for (int i = 0; i < 8; i++) y1[i] = bv;
      }
#pragma unroll
      for (int kc = 0; kc < 3; kc++) {
        OpPack bh, bl;
        int ko = kc * 32 + kbB;
        bh.q[0] = *(const v4u*)(mW1h + brow + ko);
        bl.q[0] = *(const v4u*)(mW1l + brow + ko);
        if (kc < 2) {
          bh.q[1] = *(const v4u*)(mW1h + brow + ko + 16);
          bl.q[1] = *(const v4u*)(mW1l + brow + ko + 16);
        } else {
          bh.q[1] = QZ; bl.q[1] = QZ;
        }
        y1 = __builtin_amdgcn_wmma_f32_16x16x32_bf16(false, A1[kc][0].v, false, bh.v, (short)0, y1, false, false);
        y1 = __builtin_amdgcn_wmma_f32_16x16x32_bf16(false, A1[kc][0].v, false, bl.v, (short)0, y1, false, false);
        y1 = __builtin_amdgcn_wmma_f32_16x16x32_bf16(false, A1[kc][1].v, false, bh.v, (short)0, y1, false, false);
      }
#pragma unroll
      for (int i = 0; i < 8; i++) zacc[n][i] += y0[i] * y1[i];
    }
  }

  // Signed sqrt + per-row L2 norm over the chunk's 80 columns.
  float ss[8];
#pragma unroll
  for (int i = 0; i < 8; i++) ss[i] = 0.f;
#pragma unroll
  for (int n = 0; n < 5; n++) {
#pragma unroll
    for (int i = 0; i < 8; i++) {
      float v = zacc[n][i];
      float s = (v >= 0.f) ? sqrtf(v) : -sqrtf(-v);
      zacc[n][i] = s;
      ss[i] += s * s;
    }
  }
  float inv[8];
#pragma unroll
  for (int i = 0; i < 8; i++) {
    float s = ss[i];
    // Reduce across the 16 lanes of each half (masks < 16 stay within halves).
    s += __shfl_xor(s, 1, 32);
    s += __shfl_xor(s, 2, 32);
    s += __shfl_xor(s, 4, 32);
    s += __shfl_xor(s, 8, 32);
    inv[i] = 1.0f / fmaxf(sqrtf(s), 1e-12f);
  }
#pragma unroll
  for (int n = 0; n < 5; n++) {
    int col = c * CSIZE + n * 16 + lrow;
#pragma unroll
    for (int i = 0; i < 8; i++) {
      int row = rowbase + hsel * 8 + i;
      float v = zacc[n][i] * inv[i];
      size_t off = (size_t)row * MM_SZ + col;
      unsigned short h = f2bf(v);
      zh[off] = h;
      zl[off] = f2bf(v - bf2f(h));
    }
  }
}

// ---------------------------------------------------------------------------
// Host-side launcher
// ---------------------------------------------------------------------------
extern "C" void kernel_launch(void* const* d_in, const int* in_sizes, int n_in,
                              void* d_out, int out_size, void* d_ws, size_t ws_size,
                              hipStream_t stream) {
  (void)in_sizes; (void)n_in; (void)out_size; (void)ws_size;
  const float* x0   = (const float*)d_in[0];
  const float* x1   = (const float*)d_in[1];
  const float* W0   = (const float*)d_in[2];
  const float* b0   = (const float*)d_in[3];
  const float* W1   = (const float*)d_in[4];
  const float* b1   = (const float*)d_in[5];
  const float* mW0  = (const float*)d_in[6];
  const float* mb0  = (const float*)d_in[7];
  const float* mW1  = (const float*)d_in[8];
  const float* mb1  = (const float*)d_in[9];
  const float* Wout = (const float*)d_in[10];
  const float* bout = (const float*)d_in[11];
  float* out = (float*)d_out;

  char* p = (char*)d_ws;
  auto alloc16 = [&](size_t elems) -> unsigned short* {
    unsigned short* r = (unsigned short*)p;
    p += ((elems * 2 + 255) & ~(size_t)255);
    return r;
  };
  const size_t nX    = (size_t)B_SZ * D_SZ;
  const size_t nW    = (size_t)MM_SZ * D_SZ;
  const size_t nmW   = (size_t)NCHUNK * TDIM * CSIZE;
  const size_t nWout = (size_t)OUT_N * MM_SZ;
  const size_t nH    = (size_t)B_SZ * MM_SZ;

  unsigned short *x0h = alloc16(nX),  *x0l = alloc16(nX);
  unsigned short *x1h = alloc16(nX),  *x1l = alloc16(nX);
  unsigned short *W0h = alloc16(nW),  *W0l = alloc16(nW);
  unsigned short *W1h = alloc16(nW),  *W1l = alloc16(nW);
  unsigned short *m0h = alloc16(nmW), *m0l = alloc16(nmW);
  unsigned short *m1h = alloc16(nmW), *m1l = alloc16(nmW);
  unsigned short *Woh = alloc16(nWout), *Wol = alloc16(nWout);
  unsigned short *h0h = alloc16(nH),  *h0l = alloc16(nH);
  unsigned short *h1h = alloc16(nH),  *h1l = alloc16(nH);
  unsigned short *zh  = alloc16(nH),  *zl  = alloc16(nH);

  auto split = [&](const float* src, unsigned short* hi, unsigned short* lo, size_t n) {
    int n4 = (int)(n / 4);
    k_split<<<(n4 + 255) / 256, 256, 0, stream>>>(src, hi, lo, n4);
  };
  split(x0, x0h, x0l, nX);
  split(x1, x1h, x1l, nX);
  split(W0, W0h, W0l, nW);
  split(W1, W1h, W1l, nW);
  split(mW0, m0h, m0l, nmW);
  split(mW1, m1h, m1l, nmW);
  split(Wout, Woh, Wol, nWout);

  // Stage 1: h0/h1 = x @ W^T + b  (4096 x 1600, K=2048), split-bf16 output
  dim3 g1((MM_SZ + 127) / 128, B_SZ / 128);
  k_gemm<1><<<g1, 256, 0, stream>>>(x0h, x0l, W0h, W0l, b0,
                                    nullptr, h0h, h0l, MM_SZ, D_SZ);
  k_gemm<1><<<g1, 256, 0, stream>>>(x1h, x1l, W1h, W1l, b1,
                                    nullptr, h1h, h1l, MM_SZ, D_SZ);

  // Stage 2: fused per-chunk bilinear + signed sqrt + L2 normalize -> z
  dim3 g2(B_SZ / 128, NCHUNK);
  k_bilinear<<<g2, 256, 0, stream>>>(h0h, h0l, h1h, h1l,
                                     m0h, m0l, mb0, m1h, m1l, mb1, zh, zl);

  // Stage 3: out = z @ Wout^T + bout  (4096 x 3000, K=1600), fp32 output
  dim3 g3((OUT_N + 127) / 128, B_SZ / 128);
  k_gemm<0><<<g3, 256, 0, stream>>>(zh, zl, Woh, Wol, bout,
                                    out, nullptr, nullptr, OUT_N, MM_SZ);
}
